// Conv2DCaps_38680475467929
// MI455X (gfx1250) — compile-verified, matrix operand order checked by reading, not run on hardware
//
#include <hip/hip_runtime.h>
#include <hip/hip_bf16.h>

typedef int v8i __attribute__((ext_vector_type(8)));

#define EPS 1e-8f
#define BATCH 64
#define CCH   256                 // conv channels = 32*8
#define HH    32
#define WW    32
#define PH    34                  // padded spatial (1-pixel halo)
#define PW    34
#define NPIX  (HH*WW)             // 1024
#define PPIX  (PH*PW)             // 1156
#define MTOT  (BATCH*NPIX)        // 65536 GEMM rows
#define KPC   9                   // 3x3 taps

// ---------------------------------------------------------------------------
// Prep 1: x (f32, NCHW) -> sign(x) (int8) in halo-padded NHWC layout:
//   sx[b][yy][xx][c], yy,xx in [0,34); border = 0 (conv zero padding).
//   Interior pixel (y,x) lives at (y+1, x+1).
// ---------------------------------------------------------------------------
__global__ __launch_bounds__(256) void prep_sign_x_pad(const float* __restrict__ x,
                                                       signed char* __restrict__ sx) {
    int p = blockIdx.x;          // padded pixel 0 .. 64*1156-1
    int c = threadIdx.x;         // channel 0..255
    int b   = p / PPIX;
    int rem = p - b * PPIX;
    int yy  = rem / PW;
    int xx  = rem - yy * PW;
    signed char s = 0;
    if (yy >= 1 && yy <= HH && xx >= 1 && xx <= WW) {
        float v = x[(((size_t)b * CCH + c) << 10) + ((yy - 1) << 5) + (xx - 1)];
        s = (signed char)((v > 0.0f) - (v < 0.0f));
    }
    sx[((size_t)p << 8) + c] = s;
}

// ---------------------------------------------------------------------------
// Prep 2: weights (f32, flat OIHW = o*2304 + i*9 + kk) ->
//   wt[kk][o][i] = sign(w)  (int8, N-major-K tiles for B fragments)
//   scale[o]     = mean(|w|) over 2304
// ---------------------------------------------------------------------------
__global__ __launch_bounds__(256) void prep_w(const float* __restrict__ w,
                                              signed char* __restrict__ wt,
                                              float* __restrict__ scale) {
    int o = blockIdx.x;
    int i = threadIdx.x;
    __shared__ float red[256];
    const float* wo = w + (size_t)o * (CCH * KPC);
    float s = 0.0f;
#pragma unroll
    for (int kk = 0; kk < KPC; ++kk) {
        float v = wo[i * KPC + kk];
        s += fabsf(v);
        wt[(((size_t)(kk * CCH + o)) << 8) + i] = (signed char)((v > 0.0f) - (v < 0.0f));
    }
    red[i] = s;
    __syncthreads();
    for (int st = 128; st > 0; st >>= 1) {
        if (i < st) red[i] += red[i + st];
        __syncthreads();
    }
    if (i == 0) scale[o] = red[0] / (float)(CCH * KPC);
}

// ---------------------------------------------------------------------------
// Fragment load: ISA 8-bit 16x64 layout. Per-lane base pointer already
// includes the (lane>=16 ? +8 : 0) K-half offset; VGPR pair j holds K bytes
// [16j, 16j+8).  `off` is a compile-time immediate under full unroll.
// ---------------------------------------------------------------------------
__device__ __forceinline__ v8i load_frag(const signed char* __restrict__ p, int off) {
    v8i f;
#pragma unroll
    for (int j = 0; j < 4; ++j) {
        long long v = *(const long long*)(p + off + j * 16);
        f[2 * j]     = (int)(v);
        f[2 * j + 1] = (int)(v >> 32);
    }
    return f;
}

// ---------------------------------------------------------------------------
// Implicit-GEMM binary conv + fused capsule squash + residual.
// Block: 256 threads = 8 waves (4 M-subtiles x 2 N-subtiles).
// Wave tile: 32(M) x 32(N) -> 4 accumulator frags.
// Block tile: 128(M) x 64(N).  Grid: (65536/128, 256/64) = (512, 4).
// K loop: 9 taps x 4 chunks of 64, fully unrolled; all load offsets are
// immediates off 4 per-lane base pointers.
// ---------------------------------------------------------------------------
__global__ __launch_bounds__(256) void bconv_gemm_squash(
    const signed char* __restrict__ sx, const signed char* __restrict__ wt,
    const float* __restrict__ scale, const float* __restrict__ xin,
    float* __restrict__ out) {
    int lane = threadIdx.x & 31;
    int wid  = threadIdx.x >> 5;
    int mw   = wid & 3;
    int nw   = wid >> 2;

    int Mbase = blockIdx.x * 128 + mw * 32;
    int Nbase = blockIdx.y * 64 + nw * 32;

    int half8 = (lane >> 4) << 3;   // K-half byte offset for this lane

    // Per-lane A base pointers (tap dy=dx=0 of each lane's pixel, padded coords)
    int m0 = Mbase + (lane & 15);
    int m1 = m0 + 16;
    int b0 = m0 >> 10, r0 = m0 & 1023;
    int b1 = m1 >> 10, r1 = m1 & 1023;
    const signed char* pa0 =
        sx + ((((size_t)b0 * PH + (r0 >> 5)) * PW + (r0 & 31)) << 8) + half8;
    const signed char* pa1 =
        sx + ((((size_t)b1 * PH + (r1 >> 5)) * PW + (r1 & 31)) << 8) + half8;

    // Per-lane B base pointers (kk = 0)
    const signed char* pb0 = wt + (((size_t)(Nbase      + (lane & 15))) << 8) + half8;
    const signed char* pb1 = wt + (((size_t)(Nbase + 16 + (lane & 15))) << 8) + half8;

    v8i acc00 = {0,0,0,0,0,0,0,0};
    v8i acc01 = {0,0,0,0,0,0,0,0};
    v8i acc10 = {0,0,0,0,0,0,0,0};
    v8i acc11 = {0,0,0,0,0,0,0,0};

#pragma unroll
    for (int t = 0; t < KPC * 4; ++t) {
        const int kk = t >> 2;            // tap index 0..8
        const int ch = t & 3;             // 64-wide K chunk 0..3
        const int dy = kk / 3, dx = kk % 3;
        const int aoff = ((dy * PW + dx) << 8) + (ch << 6);   // immediate
        const int boff = (kk << 16) + (ch << 6);              // immediate
        v8i a0 = load_frag(pa0, aoff);
        v8i a1 = load_frag(pa1, aoff);
        v8i b0v = load_frag(pb0, boff);
        v8i b1v = load_frag(pb1, boff);
        acc00 = __builtin_amdgcn_wmma_i32_16x16x64_iu8(true, a0, true, b0v, acc00, false, false);
        acc01 = __builtin_amdgcn_wmma_i32_16x16x64_iu8(true, a0, true, b1v, acc01, false, false);
        acc10 = __builtin_amdgcn_wmma_i32_16x16x64_iu8(true, a1, true, b0v, acc10, false, false);
        acc11 = __builtin_amdgcn_wmma_i32_16x16x64_iu8(true, a1, true, b1v, acc11, false, false);
    }

    // Epilogue: scale to f32, capsule squash (norm over 8-channel groups = 8
    // adjacent lanes inside each 16-wide N tile), residual add, store NCHW.
    float sc0 = scale[Nbase + (lane & 15)];
    float sc1 = scale[Nbase + 16 + (lane & 15)];
#pragma unroll
    for (int mi = 0; mi < 2; ++mi) {
#pragma unroll
        for (int ni = 0; ni < 2; ++ni) {
            v8i acc = (mi == 0) ? ((ni == 0) ? acc00 : acc01)
                                : ((ni == 0) ? acc10 : acc11);
            float sc = (ni == 0) ? sc0 : sc1;
            int n = Nbase + ni * 16 + (lane & 15);
#pragma unroll
            for (int r = 0; r < 8; ++r) {
                float val = sc * (float)acc[r];
                float s = val * val;
                s += __shfl_xor(s, 1, 32);
                s += __shfl_xor(s, 2, 32);
                s += __shfl_xor(s, 4, 32);
                float norm   = sqrtf(s);
                float factor = norm / (1.0f + s + EPS);
                int m  = Mbase + mi * 16 + r + ((lane >> 4) << 3);
                int b  = m >> 10;
                int pp = m & 1023;
                size_t idx = (((size_t)b * CCH + n) << 10) + pp;
                out[idx] = factor * val + xin[idx];
            }
        }
    }
}

// ---------------------------------------------------------------------------
extern "C" void kernel_launch(void* const* d_in, const int* in_sizes, int n_in,
                              void* d_out, int out_size, void* d_ws, size_t ws_size,
                              hipStream_t stream) {
    const float* x = (const float*)d_in[0];   // (64,32,8,32,32) f32
    const float* w = (const float*)d_in[1];   // (256*256*9, 1)  f32
    float* out = (float*)d_out;               // (64,32,8,32,32) f32

    char* ws = (char*)d_ws;
    signed char* sx = (signed char*)ws;                           // 18,939,904 B (padded)
    signed char* wt = (signed char*)(ws + 18939904);              //    589,824 B
    float* scale    = (float*)(ws + 18939904 + 589824);           //      1,024 B

    prep_sign_x_pad<<<dim3(BATCH * PPIX), 256, 0, stream>>>(x, sx);
    prep_w<<<dim3(CCH), 256, 0, stream>>>(w, wt, scale);

    dim3 grid(MTOT / 128, CCH / 64);
    bconv_gemm_squash<<<grid, 256, 0, stream>>>(sx, wt, scale, x, out);
}
